// GAT_87041807220900
// MI455X (gfx1250) — matrix operand order repaired and checked by reading, not compiled, not run on hardware
//
#include <hip/hip_runtime.h>
#include <hip/hip_bf16.h>

typedef __attribute__((ext_vector_type(16))) __bf16 v16bf;
typedef __attribute__((ext_vector_type(8)))  __bf16 v8bf;
typedef __attribute__((ext_vector_type(4)))  __bf16 v4bf;
typedef __attribute__((ext_vector_type(8)))  float  v8f;
typedef __attribute__((ext_vector_type(4)))  float  v4f;

__device__ __forceinline__ v16bf cat8(v8bf lo, v8bf hi) {
  return __builtin_shufflevector(lo, hi, 0,1,2,3,4,5,6,7,8,9,10,11,12,13,14,15);
}

// ---------------- utility kernels ----------------

__global__ void cast_f32_bf16(const float* __restrict__ in, __bf16* __restrict__ out, int n) {
  int i = blockIdx.x * 256 + threadIdx.x;
  if (i < n) out[i] = (__bf16)in[i];
}

// pack adjacency (int 0/1) into bitmask, one bit per edge, via wave32 ballot
__global__ void pack_adj(const int* __restrict__ adj, unsigned* __restrict__ bits) {
  int i = blockIdx.x * 256 + threadIdx.x;
  unsigned m = __builtin_amdgcn_ballot_w32(adj[i] != 0);
  if ((threadIdx.x & 31) == 0) bits[i >> 5] = m;
}

// s_src[h][i] = sum_j gT[h][j][i]*a[h][j],  s_dst with a[h][256+j]
__global__ void sdot_kernel(const __bf16* __restrict__ gT, const float* __restrict__ avec,
                            float* __restrict__ ssrc, float* __restrict__ sdst) {
  const int h = blockIdx.y;
  const int i = blockIdx.x * 256 + threadIdx.x;
  __shared__ float as[256], ad[256];
  const float* a = avec + h * 512;
  as[threadIdx.x] = a[threadIdx.x];
  ad[threadIdx.x] = a[256 + threadIdx.x];
  __syncthreads();
  const __bf16* g = gT + ((long)h << 20) + i;   // h*256*4096
  float s1 = 0.f, s2 = 0.f;
  for (int j = 0; j < 256; ++j) {
    float gv = (float)g[(long)j << 12];
    s1 += gv * as[j];
    s2 += gv * ad[j];
  }
  ssrc[(h << 12) + i] = s1;
  sdst[(h << 12) + i] = s2;
}

// masked per-row max of s_dst (per head); lrelu is monotone so row max of e is lrelu(s_src+max)
__global__ void rowmax_kernel(const unsigned* __restrict__ bits, const float* __restrict__ sdst,
                              float* __restrict__ rmax) {
  __shared__ float red[128];
  const int row = blockIdx.x, t = threadIdx.x;
  const unsigned* w = bits + (row << 7);
  for (int h = 0; h < 4; ++h) {
    const float* sd = sdst + (h << 12);
    float mx = -3.0e38f;
    for (int it = 0; it < 32; ++it) {
      int j = t + it * 128;
      unsigned b = w[j >> 5];
      if ((b >> (j & 31)) & 1u) mx = fmaxf(mx, sd[j]);
    }
    red[t] = mx;
    __syncthreads();
    for (int s = 64; s > 0; s >>= 1) {
      if (t < s) red[t] = fmaxf(red[t], red[t + s]);
      __syncthreads();
    }
    if (t == 0) rmax[(h << 12) + row] = red[0];
    __syncthreads();
  }
}

// ---------------- bf16 NT GEMM: C[M,N] = A[M,K] * B[N,K]^T ----------------
// Software-pipelined: fragments for step k+1 are loaded while step k's WMMAs execute.
// MODE 0: bf16 row-major out + bias ; MODE 1: f32 row-major out + bias
// MODE 2: bf16 TRANSPOSED out (out[col*ldo + m]) -- used to produce gT
template<int MODE, int NT>
__global__ __launch_bounds__(32) void gemm_nt(
    const __bf16* __restrict__ A, const __bf16* __restrict__ Bbase,
    const float* __restrict__ bias, void* __restrict__ Obase,
    int M, int Nn, int K, long bStride, long oStride, int ldo) {
  const int l  = threadIdx.x;
  const int r0 = l & 15, g2 = l >> 4;
  const int mbase = blockIdx.x << 4;
  const int nbase = blockIdx.y * (NT << 4);
  const __bf16* B = Bbase + (long)blockIdx.z * bStride;
  v8f zero = {0.f,0.f,0.f,0.f,0.f,0.f,0.f,0.f};
  v8f acc[NT];
#pragma unroll
  for (int t = 0; t < NT; ++t) acc[t] = zero;

  const __bf16* pa  = A + (long)(mbase + r0) * K + (g2 << 3);
  const __bf16* pbb = B + (long)(nbase + r0) * K + (g2 << 3);
  auto ldA = [&](int kb) -> v16bf {
    return cat8(*(const v8bf*)(pa + kb), *(const v8bf*)(pa + kb + 16));
  };
  auto ldB = [&](int t, int kb) -> v16bf {
    const __bf16* p = pbb + (long)(t << 4) * K + kb;
    return cat8(*(const v8bf*)p, *(const v8bf*)(p + 16));
  };

  v16bf afc = ldA(0);
  v16bf bfc[NT];
#pragma unroll
  for (int t = 0; t < NT; ++t) bfc[t] = ldB(t, 0);

#pragma unroll 2
  for (int kb = 32; kb < K; kb += 32) {
    v16bf afn = ldA(kb);            // next-step loads issue first ...
    v16bf bfn[NT];
#pragma unroll
    for (int t = 0; t < NT; ++t) bfn[t] = ldB(t, kb);
#pragma unroll
    for (int t = 0; t < NT; ++t)    // ... and overlap with these WMMAs
      acc[t] = __builtin_amdgcn_wmma_f32_16x16x32_bf16(false, afc, false, bfc[t],
                                                       (short)0, acc[t], false, false);
    afc = afn;
#pragma unroll
    for (int t = 0; t < NT; ++t) bfc[t] = bfn[t];
  }
#pragma unroll
  for (int t = 0; t < NT; ++t)
    acc[t] = __builtin_amdgcn_wmma_f32_16x16x32_bf16(false, afc, false, bfc[t],
                                                     (short)0, acc[t], false, false);

#pragma unroll
  for (int t = 0; t < NT; ++t) {
    const int col = nbase + (t << 4) + r0;
    float bv = 0.f;
    if (MODE != 2) bv = bias[col];
#pragma unroll
    for (int r = 0; r < 8; ++r) {
      const int m = mbase + (g2 << 3) + r;
      const float v = acc[t][r] + bv;
      if (MODE == 0)      ((__bf16*)Obase)[(long)m * ldo + col] = (__bf16)v;
      else if (MODE == 1) ((float*)Obase)[(long)m * ldo + col] = v;
      else ((__bf16*)Obase)[(long)blockIdx.z * oStride + (long)col * ldo + m] = (__bf16)v;
    }
  }
}

// ------- fused masked-softmax attention: attout[:, h*256:(h+1)*256] = elu(softmax(e)@g) -------
// 4-wave block per (head, 16-row tile). All 128 threads cooperatively build the 16x32 bf16
// P = exp(e-m)*mask tile in LDS (double buffered); each wave consumes the shared A-fragment
// via 2x ds_load_b128 and owns a 64-column slice of HID (4 accumulators, no spills).
// B-fragment global loads are issued BEFORE the barrier so the barrier wait hides their
// latency; next step's mask/s_dst are prefetched during the MMA phase.
__global__ __launch_bounds__(128) void attn_kernel(
    const float* __restrict__ ssrc, const float* __restrict__ sdst,
    const float* __restrict__ rmax, const unsigned* __restrict__ bits,
    const __bf16* __restrict__ gT, __bf16* __restrict__ attout) {
  const int tid = threadIdx.x;            // 0..127
  const int l   = tid & 31;               // lane in wave
  const int wv  = tid >> 5;               // wave id 0..3 -> 64-col slice
  const int r0  = l & 15, g2 = l >> 4;
  const int h = blockIdx.y;
  const int mbase = blockIdx.x << 4;

  // producer role: this thread computes P[prow][pk0..pk0+3] each K-step
  const int prow = tid >> 3;              // 0..15
  const int pk0  = (tid & 7) << 2;        // 0,4,...,28
  const int grow = mbase + prow;
  const float sr = ssrc[(h << 12) + grow];
  const float xm = sr + rmax[(h << 12) + grow];
  const float mrow = fmaxf(xm, 0.2f * xm);            // leaky relu (monotone -> row max)
  const unsigned* wr = bits + (grow << 7);
  const float* sdp = sdst + (h << 12) + pk0;

  __shared__ __align__(16) __bf16 Pbuf[2][16][32];    // 2 KB, ping-pong
  __shared__ float denom[16];

  const __bf16* gTh = gT + ((long)h << 20);
  v8f zero = {0.f,0.f,0.f,0.f,0.f,0.f,0.f,0.f};
  v8f acc[4];
#pragma unroll
  for (int nn = 0; nn < 4; ++nn) acc[nn] = zero;
  float dpart = 0.f;

  unsigned w = wr[0];
  v4f sv = *(const v4f*)(sdp);
  for (int t = 0; t < 128; ++t) {
    const int kb = t << 5;
    // ---- produce 4 elements of P ----
    v4bf pq;
#pragma unroll
    for (int i = 0; i < 4; ++i) {
      const float x = sr + sv[i];
      const float e = fmaxf(x, 0.2f * x);             // leaky relu
      const float pv = ((w >> (pk0 + i)) & 1u) ? __expf(e - mrow) : 0.f;
      dpart += pv;
      pq[i] = (__bf16)pv;
    }
    *(v4bf*)(&Pbuf[t & 1][prow][pk0]) = pq;
    // ---- issue this step's B-fragment loads before the barrier (latency hidden) ----
    const __bf16* pb0 = gTh + (long)((wv << 6) + r0) * 4096 + kb + (g2 << 3);
    v8bf b0[4], b1[4];
#pragma unroll
    for (int nn = 0; nn < 4; ++nn) {
      const __bf16* pb = pb0 + ((long)nn << 16);      // nn*16*4096
      b0[nn] = *(const v8bf*)pb;
      b1[nn] = *(const v8bf*)(pb + 16);
    }
    __syncthreads();
    // ---- prefetch next step's mask word + s_dst vector ----
    if (t < 127) {
      w  = wr[t + 1];
      sv = *(const v4f*)(sdp + kb + 32);
    }
    // ---- shared A fragment + 4 WMMAs ----
    const __bf16* pp = &Pbuf[t & 1][r0][g2 << 3];
    v16bf af = cat8(*(const v8bf*)pp, *(const v8bf*)(pp + 16));
#pragma unroll
    for (int nn = 0; nn < 4; ++nn)
      acc[nn] = __builtin_amdgcn_wmma_f32_16x16x32_bf16(false, af, false, cat8(b0[nn], b1[nn]),
                                                        (short)0, acc[nn], false, false);
  }

  // ---- row denominators: reduce 8 producer threads per row ----
  dpart += __shfl_xor(dpart, 1, 32);
  dpart += __shfl_xor(dpart, 2, 32);
  dpart += __shfl_xor(dpart, 4, 32);
  if ((tid & 7) == 0) denom[prow] = dpart;
  __syncthreads();

  float dinv[8];
#pragma unroll
  for (int r = 0; r < 8; ++r) dinv[r] = 1.f / denom[(g2 << 3) + r];
#pragma unroll
  for (int nn = 0; nn < 4; ++nn) {
    const int col = (h << 8) + (wv << 6) + (nn << 4) + r0;
#pragma unroll
    for (int r = 0; r < 8; ++r) {
      const int m = mbase + (g2 << 3) + r;
      float v = acc[nn][r] * dinv[r];
      v = (v > 0.f) ? v : (__expf(v) - 1.f);          // elu
      attout[((long)m << 10) + col] = (__bf16)v;
    }
  }
}

// ---------------- launch ----------------

extern "C" void kernel_launch(void* const* d_in, const int* in_sizes, int n_in,
                              void* d_out, int out_size, void* d_ws, size_t ws_size,
                              hipStream_t stream) {
  (void)in_sizes; (void)n_in; (void)out_size; (void)ws_size;
  const float* label  = (const float*)d_in[0];
  const float* W1     = (const float*)d_in[1];
  const float* a1     = (const float*)d_in[2];
  const float* W2     = (const float*)d_in[3];
  const float* a2     = (const float*)d_in[4];
  const float* out_w  = (const float*)d_in[5];
  const float* out_b  = (const float*)d_in[6];
  const float* out2_w = (const float*)d_in[7];
  const float* out2_b = (const float*)d_in[8];
  const int*   adj    = (const int*)d_in[9];

  char* base = (char*)d_ws;
  size_t off = 0;
  auto alloc = [&](size_t bytes) {
    char* p = base + off;
    off = (off + bytes + 255) & ~(size_t)255;
    return p;
  };
  __bf16* xbf     = (__bf16*)  alloc((size_t)4096 * 768 * 2);
  __bf16* w1bf    = (__bf16*)  alloc((size_t)4 * 256 * 768 * 2);
  __bf16* w2bf    = (__bf16*)  alloc((size_t)4 * 256 * 768 * 2);
  __bf16* outwbf  = (__bf16*)  alloc((size_t)768 * 1024 * 2);
  __bf16* out2wbf = (__bf16*)  alloc((size_t)32 * 1024 * 2);
  unsigned* adjb  = (unsigned*)alloc((size_t)4096 * 128 * 4);
  __bf16* gT      = (__bf16*)  alloc((size_t)4 * 256 * 4096 * 2);
  float*  ssrc    = (float*)   alloc((size_t)4 * 4096 * 4);
  float*  sdst    = (float*)   alloc((size_t)4 * 4096 * 4);
  float*  rmax    = (float*)   alloc((size_t)4 * 4096 * 4);
  __bf16* attout  = (__bf16*)  alloc((size_t)4096 * 1024 * 2);
  __bf16* x1bf    = (__bf16*)  alloc((size_t)4096 * 768 * 2);

  // preprocessing
  pack_adj<<<dim3((4096 * 4096) / 256), 256, 0, stream>>>(adj, adjb);
  cast_f32_bf16<<<dim3((4096 * 768) / 256), 256, 0, stream>>>(label, xbf, 4096 * 768);
  cast_f32_bf16<<<dim3((4 * 256 * 768) / 256), 256, 0, stream>>>(W1, w1bf, 4 * 256 * 768);
  cast_f32_bf16<<<dim3((4 * 256 * 768) / 256), 256, 0, stream>>>(W2, w2bf, 4 * 256 * 768);
  cast_f32_bf16<<<dim3((768 * 1024) / 256), 256, 0, stream>>>(out_w, outwbf, 768 * 1024);
  cast_f32_bf16<<<dim3((32 * 1024) / 256), 256, 0, stream>>>(out2_w, out2wbf, 32 * 1024);

  // ---- layer 1 ----
  gemm_nt<2, 4><<<dim3(256, 4, 4), 32, 0, stream>>>(xbf, w1bf, nullptr, gT,
      4096, 256, 768, 256l * 768, 256l * 4096, 4096);                 // gT[h] = (x @ W1[h]^T)^T
  sdot_kernel<<<dim3(16, 4), 256, 0, stream>>>(gT, a1, ssrc, sdst);
  rowmax_kernel<<<dim3(4096), 128, 0, stream>>>(adjb, sdst, rmax);
  attn_kernel<<<dim3(256, 4), 128, 0, stream>>>(ssrc, sdst, rmax, adjb, gT, attout);
  gemm_nt<0, 4><<<dim3(256, 12, 1), 32, 0, stream>>>(attout, outwbf, out_b, x1bf,
      4096, 768, 1024, 0, 0, 768);                                    // x1 = attout @ out_w^T + b

  // ---- layer 2 ----
  gemm_nt<2, 4><<<dim3(256, 4, 4), 32, 0, stream>>>(x1bf, w2bf, nullptr, gT,
      4096, 256, 768, 256l * 768, 256l * 4096, 4096);
  sdot_kernel<<<dim3(16, 4), 256, 0, stream>>>(gT, a2, ssrc, sdst);
  rowmax_kernel<<<dim3(4096), 128, 0, stream>>>(adjb, sdst, rmax);
  attn_kernel<<<dim3(256, 4), 128, 0, stream>>>(ssrc, sdst, rmax, adjb, gT, attout);
  gemm_nt<1, 2><<<dim3(256, 1, 1), 32, 0, stream>>>(attout, out2wbf, out2_b, d_out,
      4096, 32, 1024, 0, 0, 32);                                      // final f32 output
}
